// VQCodebook_25142738551443
// MI455X (gfx1250) — matrix operand order repaired and checked by reading, not compiled, not run on hardware
//
#include <hip/hip_runtime.h>

// ---------------- problem constants ----------------
#define NROWS   131072     // N_GROUPS
#define GS      128        // GROUP_SIZE (K-dim of dot products)
#define NCODES  2048       // CODEBOOK_SIZE

// ---------------- tiling ----------------
#define M_PER_WAVE  32     // x rows per wave (two 16-row WMMA groups)
#define WAVES       8      // 256 threads
#define M_TILE      (M_PER_WAVE * WAVES)   // 256 rows per block
#define CHUNK       128    // codebook rows staged in LDS per iteration
#define NT          (CHUNK / 16)
#define LDS_ROW     136    // ushorts per LDS codebook row: 128 data + 8 pad (272B stride -> 4-bank skew)

// ---------------- vector types (trivial, union-friendly) ----------------
typedef __attribute__((ext_vector_type(4)))  float         f4;
typedef __attribute__((ext_vector_type(8)))  float         v8f;
typedef __attribute__((ext_vector_type(4)))  unsigned int  u4;
typedef __attribute__((ext_vector_type(2)))  unsigned int  u2;
typedef __attribute__((ext_vector_type(16))) __bf16        v16bf;

union FragBF {
    v16bf v;
    unsigned short s[16];
    u4 q[2];
};

// float -> bf16, round-to-nearest-even (integer-only, always compiles)
static __device__ __forceinline__ unsigned short f2bf(float f) {
    unsigned int u = __float_as_uint(f);
    u += 0x7FFFu + ((u >> 16) & 1u);
    return (unsigned short)(u >> 16);
}

// ============================================================
// Kernel 1: codebook f32 -> bf16 (ws) + c_sq (ws)
// grid 256 x 256 threads: wave per codebook row (8 rows/block)
// ============================================================
__global__ void vq_prep_kernel(const float* __restrict__ codebook,
                               unsigned short* __restrict__ cb_bf16,
                               float* __restrict__ c_sq) {
    const int tid  = threadIdx.x;
    const int wave = tid >> 5;
    const int lane = tid & 31;
    const int row  = blockIdx.x * 8 + wave;   // 256 blocks * 8 = 2048 rows

    const f4* crow = reinterpret_cast<const f4*>(codebook + (size_t)row * GS);
    f4 v = crow[lane];                         // 4 floats per lane

    unsigned short h0 = f2bf(v[0]), h1 = f2bf(v[1]), h2 = f2bf(v[2]), h3 = f2bf(v[3]);
    u2 packed;
    packed[0] = (unsigned int)h0 | ((unsigned int)h1 << 16);
    packed[1] = (unsigned int)h2 | ((unsigned int)h3 << 16);
    *reinterpret_cast<u2*>(cb_bf16 + (size_t)row * GS + lane * 4) = packed;

    float sq = v[0]*v[0] + v[1]*v[1] + v[2]*v[2] + v[3]*v[3];
    #pragma unroll
    for (int off = 16; off >= 1; off >>= 1) sq += __shfl_xor(sq, off);
    if (lane == 0) c_sq[row] = sq;
}

// ============================================================
// Kernel 2: fused distance-GEMM + argmin + gather + loss partials
// grid 512 x 256 threads. Wave w owns rows block*256 + w*32 .. +31
// (two 16-row WMMA groups sharing each B fragment -> 1 WMMA per ds_load_b128)
// ============================================================
__global__ void __launch_bounds__(256)
vq_main_kernel(const float* __restrict__ x,
               const float* __restrict__ codebook,
               const unsigned short* __restrict__ cb_bf16,
               const float* __restrict__ c_sq,
               float* __restrict__ out_q,      // [NROWS][GS]
               int*   __restrict__ out_idx,    // [NROWS]
               float* __restrict__ partials) { // [gridDim.x]
    __shared__ unsigned short tile[CHUNK * LDS_ROW]; // 34816 B, 4-bank row skew
    __shared__ float csq_s[CHUNK];
    __shared__ int   idx_s[WAVES * M_PER_WAVE];
    __shared__ float loss_s[WAVES];

    const int tid  = threadIdx.x;
    const int wave = tid >> 5;
    const int lane = tid & 31;
    const int lo   = lane & 15;       // column / row-in-tile selector
    const int hi   = lane >> 4;       // half-wave selector
    const int blockRow0 = blockIdx.x * M_TILE;

    // ---- build 2 x four 16x32 bf16 A fragments for this wave's 32 rows ----
    // ISA layout: lanes 0-15 -> M=lane, K = kk*32 + {0..7, 16..23}
    //             lanes 16-31 -> M=lane-16, K = kk*32 + {8..15, 24..31}
    FragBF afrag[2][4];
    #pragma unroll
    for (int g = 0; g < 2; ++g) {
        const int myRow = blockRow0 + wave * M_PER_WAVE + g * 16 + lo;
        const float* xr = x + (size_t)myRow * GS;
        #pragma unroll
        for (int kk = 0; kk < 4; ++kk) {
            const int kb = kk * 32 + hi * 8;
            f4 c0 = *reinterpret_cast<const f4*>(xr + kb);
            f4 c1 = *reinterpret_cast<const f4*>(xr + kb + 4);
            f4 c2 = *reinterpret_cast<const f4*>(xr + kb + 16);
            f4 c3 = *reinterpret_cast<const f4*>(xr + kb + 20);
            #pragma unroll
            for (int i = 0; i < 4; ++i) {
                afrag[g][kk].s[i]      = f2bf(c0[i]);
                afrag[g][kk].s[4 + i]  = f2bf(c1[i]);
                afrag[g][kk].s[8 + i]  = f2bf(c2[i]);
                afrag[g][kk].s[12 + i] = f2bf(c3[i]);
            }
        }
    }

    float minv[2][8];
    int   mini[2][8];
    #pragma unroll
    for (int g = 0; g < 2; ++g)
        #pragma unroll
        for (int v = 0; v < 8; ++v) { minv[g][v] = 3.402823466e38f; mini[g][v] = 0; }

    // ---- stream codebook through LDS in 128-code chunks ----
    for (int chunkBase = 0; chunkBase < NCODES; chunkBase += CHUNK) {
        __syncthreads();  // previous chunk fully consumed
        for (int i = tid; i < CHUNK * 16; i += 256) {   // 16-byte segments
            const int r = i >> 4, s = i & 15;
            u4 d = *reinterpret_cast<const u4*>(cb_bf16 + (size_t)(chunkBase + r) * GS + s * 8);
            *reinterpret_cast<u4*>(&tile[r * LDS_ROW + s * 8]) = d;
        }
        for (int i = tid; i < CHUNK; i += 256) csq_s[i] = c_sq[chunkBase + i];
        __syncthreads();

        for (int nt = 0; nt < NT; ++nt) {
            // hoist all four B fragments (8 x ds_load_b128) so loads pipeline
            // B 32x16 layout: lane -> column N=lo; lanes 0-15 hold K=kk*32+{0..7,16..23},
            // lanes 16-31 hold K=kk*32+{8..15,24..31}
            FragBF b[4];
            const unsigned short* brow = &tile[(nt * 16 + lo) * LDS_ROW + hi * 8];
            #pragma unroll
            for (int kk = 0; kk < 4; ++kk) {
                b[kk].q[0] = *reinterpret_cast<const u4*>(brow + kk * 32);
                b[kk].q[1] = *reinterpret_cast<const u4*>(brow + kk * 32 + 16);
            }

            v8f acc0 = {0.f, 0.f, 0.f, 0.f, 0.f, 0.f, 0.f, 0.f};
            v8f acc1 = {0.f, 0.f, 0.f, 0.f, 0.f, 0.f, 0.f, 0.f};
            #pragma unroll
            for (int kk = 0; kk < 4; ++kk) {
                acc0 = __builtin_amdgcn_wmma_f32_16x16x32_bf16(
                        false, afrag[0][kk].v, false, b[kk].v, (short)0, acc0, false, false);
                acc1 = __builtin_amdgcn_wmma_f32_16x16x32_bf16(
                        false, afrag[1][kk].v, false, b[kk].v, (short)0, acc1, false, false);
            }

            // score for row m = v + 8*hi (per group), code n = chunkBase + nt*16 + lo
            const float cs = csq_s[nt * 16 + lo];
            const int   n  = chunkBase + nt * 16 + lo;
            #pragma unroll
            for (int v = 0; v < 8; ++v) {
                const float s0 = cs - 2.0f * acc0[v];
                if (s0 < minv[0][v]) { minv[0][v] = s0; mini[0][v] = n; }
                const float s1 = cs - 2.0f * acc1[v];
                if (s1 < minv[1][v]) { minv[1][v] = s1; mini[1][v] = n; }
            }
        }
    }

    // ---- cross-lane argmin within each 16-lane half (first-index tie-break) ----
    #pragma unroll
    for (int off = 1; off < 16; off <<= 1) {
        #pragma unroll
        for (int g = 0; g < 2; ++g) {
            #pragma unroll
            for (int v = 0; v < 8; ++v) {
                const float ov = __shfl_xor(minv[g][v], off);
                const int   oi = __shfl_xor(mini[g][v], off);
                if (ov < minv[g][v] || (ov == minv[g][v] && oi < mini[g][v])) {
                    minv[g][v] = ov; mini[g][v] = oi;
                }
            }
        }
    }
    if (lo == 0) {
        #pragma unroll
        for (int g = 0; g < 2; ++g) {
            #pragma unroll
            for (int v = 0; v < 8; ++v) {
                const int m = wave * M_PER_WAVE + g * 16 + v + 8 * hi; // row within block
                idx_s[m] = mini[g][v];
                out_idx[blockRow0 + m] = mini[g][v];
            }
        }
    }
    __syncthreads();

    // ---- gather winning codebook rows (f32), write quantized, accumulate loss ----
    const f4* cb4 = reinterpret_cast<const f4*>(codebook);
    const f4* x4  = reinterpret_cast<const f4*>(x);
    f4*       o4  = reinterpret_cast<f4*>(out_q);

    float loss = 0.f;
    #pragma unroll 4
    for (int r = 0; r < M_PER_WAVE; ++r) {
        const int m    = wave * M_PER_WAVE + r;
        const int code = idx_s[m];
        const size_t grow = (size_t)(blockRow0 + m);
        f4 q  = cb4[(size_t)code * 32 + lane];
        f4 xv = x4[grow * 32 + lane];
        o4[grow * 32 + lane] = q;
        f4 d = xv - q;
        loss += d[0]*d[0] + d[1]*d[1] + d[2]*d[2] + d[3]*d[3];
    }
    #pragma unroll
    for (int off = 16; off >= 1; off >>= 1) loss += __shfl_xor(loss, off);
    if (lane == 0) loss_s[wave] = loss;
    __syncthreads();
    if (tid == 0) {
        float t = 0.f;
        #pragma unroll
        for (int w = 0; w < WAVES; ++w) t += loss_s[w];
        partials[blockIdx.x] = t;
    }
}

// ============================================================
// Kernel 3: deterministic reduction of block partials -> mean loss
// ============================================================
__global__ void vq_finalize_kernel(const float* __restrict__ partials, int n,
                                   float* __restrict__ out_loss) {
    __shared__ float s[256];
    const int tid = threadIdx.x;
    float t = 0.f;
    for (int i = tid; i < n; i += 256) t += partials[i];
    s[tid] = t;
    __syncthreads();
    for (int off = 128; off >= 1; off >>= 1) {
        if (tid < off) s[tid] += s[tid + off];
        __syncthreads();
    }
    if (tid == 0) out_loss[0] = s[0] / ((float)NROWS * (float)GS);
}

// ============================================================
extern "C" void kernel_launch(void* const* d_in, const int* in_sizes, int n_in,
                              void* d_out, int out_size, void* d_ws, size_t ws_size,
                              hipStream_t stream) {
    const float* x        = (const float*)d_in[0];   // [131072,128]
    const float* codebook = (const float*)d_in[1];   // [2048,128]

    float* out_q    = (float*)d_out;                         // NROWS*GS floats
    int*   out_idx  = (int*)((float*)d_out + (size_t)NROWS * GS);
    float* out_loss = (float*)d_out + (size_t)NROWS * GS + NROWS;

    // workspace layout
    unsigned short* cb_bf16 = (unsigned short*)d_ws;                   // 512 KB
    float* c_sq     = (float*)((char*)d_ws + (size_t)NCODES * GS * 2); // 8 KB
    float* partials = c_sq + NCODES;                                   // 2 KB

    const int nBlocks = NROWS / M_TILE;  // 512

    vq_prep_kernel<<<NCODES / 8, 256, 0, stream>>>(codebook, cb_bf16, c_sq);
    vq_main_kernel<<<nBlocks, 256, 0, stream>>>(x, codebook, cb_bf16, c_sq,
                                                out_q, out_idx, partials);
    vq_finalize_kernel<<<1, 256, 0, stream>>>(partials, nBlocks, out_loss);
}
